// decoder_33663953666589
// MI455X (gfx1250) — matrix operand order repaired and checked by reading, not compiled, not run on hardware
//
#include <hip/hip_runtime.h>
#include <cmath>
#include <cstdint>

// ===========================================================================
// MI455X (gfx1250, wave32) decoder + CLIP-style pooled-logits pipeline.
//
// Algebraic folding: self-attn L=1 -> softmax==1; cross-attn K/V projections
// folded through the attention math so `memory` is only ever streamed, never
// projected; final einsums fused into one batched bf16 WMMA GEMM with TDM
// tile loads.  Memory-bound: ~0.5GB HBM traffic total (~22us @ 23.3TB/s).
// ===========================================================================

typedef __bf16 bf16_t;
typedef __bf16 v16bf __attribute__((ext_vector_type(16)));
typedef float  v8f   __attribute__((ext_vector_type(8)));
typedef unsigned int u32x4 __attribute__((ext_vector_type(4)));
typedef int    i32x8 __attribute__((ext_vector_type(8)));
typedef int    i32x4 __attribute__((ext_vector_type(4)));

#ifndef __has_builtin
#define __has_builtin(x) 0
#endif
#if defined(__HIP_DEVICE_COMPILE__) && __has_builtin(__builtin_amdgcn_tensor_load_to_lds)
#define USE_TDM 1
#else
#define USE_TDM 0
#endif

static constexpr int BATCH  = 128;
static constexpr int SEQ    = 256;
static constexpr int DIN    = 768;
static constexpr int DMODEL = 1536;
static constexpr int NHEAD  = 8;
static constexpr int DHEAD  = 192;
static constexpr int DFF    = 2048;

// --------------------------- device helpers -------------------------------

__device__ __forceinline__ v8f wmma_bf16(v16bf a, v16bf b, v8f c) {
  return __builtin_amdgcn_wmma_f32_16x16x32_bf16(false, a, false, b, (short)0, c,
                                                 false, false);
}

__device__ __forceinline__ float warp_sum(float v) {
#pragma unroll
  for (int o = 16; o > 0; o >>= 1) v += __shfl_xor(v, o, 32);
  return v;
}
__device__ __forceinline__ float warp_max(float v) {
#pragma unroll
  for (int o = 16; o > 0; o >>= 1) v = fmaxf(v, __shfl_xor(v, o, 32));
  return v;
}

// A fragment (16x32 bf16) from row-major f32 A.  Layout: lanes 0-15 hold
// K=0..15; lanes 16-31 hold K=16..31 of row M=lane&15.
__device__ __forceinline__ v16bf afrag_f32(const float* A, int lda, int k0) {
  const int lane = threadIdx.x & 31;
  const float* p = A + (int64_t)(lane & 15) * lda + k0 + ((lane < 16) ? 0 : 16);
  v16bf r;
#pragma unroll
  for (int e = 0; e < 8; ++e) r[e] = (bf16_t)p[e];
#pragma unroll
  for (int e = 0; e < 8; ++e) r[8 + e] = (bf16_t)p[8 + e];
  return r;
}

// B fragment (32x16 bf16), B stored [N][K] (k contiguous), f32 source.
__device__ __forceinline__ v16bf bfrag_f32_nk(const float* B, int ldb, int k0) {
  const int lane = threadIdx.x & 31;
  const float* p = B + (int64_t)(lane & 15) * ldb + k0 + ((lane < 16) ? 0 : 16);
  v16bf r;
#pragma unroll
  for (int e = 0; e < 16; ++e) r[e] = (bf16_t)p[e];
  return r;
}
// B fragment, B stored [K][N] (n contiguous, k strided), f32 source.
__device__ __forceinline__ v16bf bfrag_f32_kn(const float* B, int ldb, int k0) {
  const int lane = threadIdx.x & 31;
  const float* p = B + (int64_t)(k0 + ((lane < 16) ? 0 : 16)) * ldb + (lane & 15);
  v16bf r;
#pragma unroll
  for (int e = 0; e < 16; ++e) r[e] = (bf16_t)p[(int64_t)e * ldb];
  return r;
}
// bf16-source variants (memory tensor).
__device__ __forceinline__ v16bf bfrag_bf_nk(const bf16_t* B, int ldb, int k0) {
  const int lane = threadIdx.x & 31;
  const bf16_t* p = B + (int64_t)(lane & 15) * ldb + k0 + ((lane < 16) ? 0 : 16);
  v16bf r;
#pragma unroll
  for (int e = 0; e < 16; ++e) r[e] = p[e];
  return r;
}
__device__ __forceinline__ v16bf bfrag_bf_kn(const bf16_t* B, int ldb, int k0) {
  const int lane = threadIdx.x & 31;
  const bf16_t* p = B + (int64_t)(k0 + ((lane < 16) ? 0 : 16)) * ldb + (lane & 15);
  v16bf r;
#pragma unroll
  for (int e = 0; e < 16; ++e) r[e] = p[(int64_t)e * ldb];
  return r;
}

// --------------------------- generic WMMA GEMM -----------------------------
// C[M,N] = alpha * A[M,K] x op(B) + bias, bf16 compute / f32 accumulate.
// M == 128 (8 waves x 16 rows per 256-thread block), grid.x = N/16.
// layoutB==0: B is [N][K] row-major (i.e. x @ W^T);  layoutB==1: B is [K][N].
__global__ __launch_bounds__(256) void gemm_wmma(
    const float* __restrict__ A, int lda,
    const float* __restrict__ B, int ldb, int layoutB,
    float* __restrict__ C, int ldc,
    const float* __restrict__ bias,
    int M, int N, int K, float alpha, int relu)
{
  const int wave = threadIdx.x >> 5;
  const int m0 = wave * 16;
  const int n0 = blockIdx.x * 16;
  if (m0 >= M || n0 >= N) return;
  const float* Aw = A + (int64_t)m0 * lda;
  v8f acc = {0.f, 0.f, 0.f, 0.f, 0.f, 0.f, 0.f, 0.f};
  for (int k0 = 0; k0 < K; k0 += 32) {
    v16bf a = afrag_f32(Aw, lda, k0);
    v16bf b = (layoutB == 0) ? bfrag_f32_nk(B + (int64_t)n0 * ldb, ldb, k0)
                             : bfrag_f32_kn(B + n0, ldb, k0);
    acc = wmma_bf16(a, b, acc);
  }
  const int lane = threadIdx.x & 31;
  const int n = lane & 15;
  const int moff = (lane < 16) ? 0 : 8;   // C tile: VGPR r -> row r (+8 hi half)
  const float bv = bias ? bias[n0 + n] : 0.f;
  float* Cw = C + (int64_t)m0 * ldc + n0;
#pragma unroll
  for (int r = 0; r < 8; ++r) {
    float v = alpha * acc[r] + bv;
    if (relu) v = fmaxf(v, 0.f);
    Cw[(int64_t)(moff + r) * ldc + n] = v;
  }
}

// ------------------- batched 16-row GEMM (per-sample CA) -------------------
// C_b[16,N] = alpha * A_b[16,K] x op(B_b);  B is bf16 (the memory tensor).
__global__ __launch_bounds__(256) void gemm16_batched(
    const float* __restrict__ A, int64_t strideA, int lda, int K,
    const bf16_t* __restrict__ B, int64_t strideB, int ldb, int layoutB,
    float* __restrict__ C, int64_t strideC, int ldc, int N, float alpha)
{
  const int b = blockIdx.x;
  const int wave = threadIdx.x >> 5;
  const int n0 = (blockIdx.y * 8 + wave) * 16;
  if (n0 >= N) return;
  const float*  Ab = A + (int64_t)b * strideA;
  const bf16_t* Bb = B + (int64_t)b * strideB;
  v8f acc = {0.f, 0.f, 0.f, 0.f, 0.f, 0.f, 0.f, 0.f};
  for (int k0 = 0; k0 < K; k0 += 32) {
    v16bf a  = afrag_f32(Ab, lda, k0);
    v16bf bb = (layoutB == 0) ? bfrag_bf_nk(Bb + (int64_t)n0 * ldb, ldb, k0)
                              : bfrag_bf_kn(Bb + n0, ldb, k0);
    acc = wmma_bf16(a, bb, acc);
  }
  const int lane = threadIdx.x & 31;
  const int n = lane & 15;
  const int moff = (lane < 16) ? 0 : 8;
  float* Cw = C + (int64_t)b * strideC + n0;
#pragma unroll
  for (int r = 0; r < 8; ++r)
    Cw[(int64_t)(moff + r) * ldc + n] = alpha * acc[r];
}

// ------------------ TDM tile loader for the flagship kernel ----------------
// Loads a 128x32 bf16 tile (row stride = rowStrideElems) into LDS.
__device__ __forceinline__ void tile_load128x32(bf16_t* ldsDst,
                                                const bf16_t* g,
                                                int rowStrideElems)
{
#if USE_TDM
  if ((threadIdx.x >> 5) == 0) {
    // D# group 0: count=1, lds_addr, global_addr[56:0], type=2 ("image").
    const uint64_t ga = (uint64_t)(uintptr_t)g;
    u32x4 g0;
    g0[0] = 1u;
    g0[1] = (uint32_t)(uintptr_t)ldsDst;             // LDS byte offset
    g0[2] = (uint32_t)ga;                            // global_addr[31:0]
    g0[3] = (uint32_t)((ga >> 32) & 0x01ffffffu) | (2u << 30);
    // D# group 1: data_size=2B, tensor_dim0=rowStride, tensor_dim1=128,
    // tile_dim0=32, tile_dim1=128, tensor_dim0_stride=rowStride.
    const uint32_t td0 = (uint32_t)rowStrideElems;
    const uint32_t td1 = 128u;
    i32x8 g1;
    g1[0] = (int)(1u << 16);                         // data_size = 1 (2 bytes)
    g1[1] = (int)((td0 & 0xffffu) << 16);            // tensor_dim0[15:0]
    g1[2] = (int)((td0 >> 16) | ((td1 & 0xffffu) << 16));
    g1[3] = (int)((td1 >> 16) | (32u << 16));        // tile_dim0 = 32
    g1[4] = (int)128u;                               // tile_dim1=128, tile_dim2=0
    g1[5] = (int)td0;                                // tensor_dim0_stride[31:0]
    g1[6] = 0;
    g1[7] = 0;
    const i32x4 gz4 = {0, 0, 0, 0};
    const i32x8 gz8 = {0, 0, 0, 0, 0, 0, 0, 0};
    // 6-arg form (clang-23 / therock-10.0): groups 0..3 + extra group + cpol.
    __builtin_amdgcn_tensor_load_to_lds(g0, g1, gz4, gz4, gz8, 0);
  }
#else
  for (int idx = threadIdx.x; idx < 128 * 32; idx += 256) {
    const int r = idx >> 5, c = idx & 31;
    ldsDst[idx] = g[(int64_t)r * rowStrideElems + c];
  }
#endif
}

__device__ __forceinline__ void tile_wait_all()
{
#if USE_TDM
  if ((threadIdx.x >> 5) == 0) __builtin_amdgcn_s_wait_tensorcnt(0);
#endif
  __syncthreads();
}

// --------------------- flagship fused batched einsum -----------------------
// For each j: C(256x256) = A2(256x1536) @ mem_bf16[j]^T.
// Rows 0..127 of A2 = q~ (pre-scaled by 1/sqrt(D)) -> scores[i,j,s];
// rows 128..255 = x_sq -> dots[i,j,s].
// Block = 256 thr (8 waves, 2x4), block tile 128x128, wave tile 64x32
// (8 WMMA accumulators).  K streamed in 32-wide double-buffered LDS tiles
// fetched by the Tensor Data Mover.
__global__ __launch_bounds__(256) void attn_scores_dots(
    const bf16_t* __restrict__ A2, const bf16_t* __restrict__ memB,
    float* __restrict__ scores, float* __restrict__ dots, int K)
{
  __shared__ bf16_t lA[2][128 * 32];
  __shared__ bf16_t lB[2][128 * 32];
  const int j    = blockIdx.y;
  const int bm   = blockIdx.x >> 1;     // 0: scores rows, 1: dot rows
  const int bn   = blockIdx.x & 1;      // s half
  const int wave = threadIdx.x >> 5;
  const int lane = threadIdx.x & 31;
  const int wm = wave >> 2, wn = wave & 3;
  const bf16_t* Ag = A2 + (int64_t)(bm * 128) * K;
  const bf16_t* Bg = memB + ((int64_t)j * 256 + bn * 128) * K;

  v8f acc[4][2];
#pragma unroll
  for (int t = 0; t < 4; ++t)
#pragma unroll
    for (int u = 0; u < 2; ++u)
      acc[t][u] = (v8f){0.f, 0.f, 0.f, 0.f, 0.f, 0.f, 0.f, 0.f};

  tile_load128x32(&lA[0][0], Ag, K);
  tile_load128x32(&lB[0][0], Bg, K);
  const int nk = K / 32;
  const int mrow = lane & 15;
  const int akb = (lane < 16) ? 0 : 16;
  const int bkb = (lane < 16) ? 0 : 16;

  for (int kt = 0; kt < nk; ++kt) {
    const int cur = kt & 1;
    tile_wait_all();
    if (kt + 1 < nk) {
      tile_load128x32(&lA[cur ^ 1][0], Ag + (kt + 1) * 32, K);
      tile_load128x32(&lB[cur ^ 1][0], Bg + (kt + 1) * 32, K);
    }
    v16bf af[4], bfm[2];
#pragma unroll
    for (int t = 0; t < 4; ++t) {
      const bf16_t* pa = &lA[cur][(wm * 64 + t * 16 + mrow) * 32 + akb];
#pragma unroll
      for (int e = 0; e < 8; ++e) af[t][e] = pa[e];
#pragma unroll
      for (int e = 0; e < 8; ++e) af[t][8 + e] = pa[8 + e];
    }
#pragma unroll
    for (int u = 0; u < 2; ++u) {
      const bf16_t* pb = &lB[cur][(wn * 32 + u * 16 + mrow) * 32 + bkb];
#pragma unroll
      for (int e = 0; e < 16; ++e) bfm[u][e] = pb[e];
    }
#pragma unroll
    for (int t = 0; t < 4; ++t)
#pragma unroll
      for (int u = 0; u < 2; ++u)
        acc[t][u] = wmma_bf16(af[t], bfm[u], acc[t][u]);
    __syncthreads();
  }

  const int moff = (lane < 16) ? 0 : 8;
  const int ncol = lane & 15;
  float* Cb = (bm == 0) ? scores : dots;
#pragma unroll
  for (int t = 0; t < 4; ++t) {
    const int i0 = wm * 64 + t * 16;
#pragma unroll
    for (int u = 0; u < 2; ++u) {
      const int s0 = bn * 128 + wn * 32 + u * 16;
#pragma unroll
      for (int r = 0; r < 8; ++r)
        Cb[((int64_t)(i0 + moff + r) * 128 + j) * 256 + s0 + ncol] = acc[t][u][r];
    }
  }
}

// --------------------------- small kernels ---------------------------------

__global__ void f32_to_bf16(const float* __restrict__ in,
                            bf16_t* __restrict__ out, int64_t n) {
  const int64_t i = (int64_t)blockIdx.x * 256 + threadIdx.x;
  if (i < n) out[i] = (bf16_t)in[i];
}

__global__ void zero_f32(float* __restrict__ p, int64_t n) {
  const int64_t i = (int64_t)blockIdx.x * 256 + threadIdx.x;
  if (i < n) p[i] = 0.f;
}

// out = LayerNorm(a + b) * w + bias;  one 256-thread block per row.
__global__ __launch_bounds__(256) void ln_residual(
    const float* __restrict__ a, const float* __restrict__ b,
    const float* __restrict__ w, const float* __restrict__ bias,
    float* __restrict__ out, int dm)
{
  const int row = blockIdx.x;
  const float* pa = a + (int64_t)row * dm;
  const float* pb = b ? b + (int64_t)row * dm : nullptr;
  float vals[8];
  const int per = dm >> 8;
  float s = 0.f, s2 = 0.f;
  for (int i = 0; i < per; ++i) {
    const int idx = threadIdx.x + (i << 8);
    const float v = pa[idx] + (pb ? pb[idx] : 0.f);
    vals[i] = v; s += v; s2 += v * v;
  }
  __shared__ float rs[8], rs2[8];
  s = warp_sum(s); s2 = warp_sum(s2);
  if ((threadIdx.x & 31) == 0) { rs[threadIdx.x >> 5] = s; rs2[threadIdx.x >> 5] = s2; }
  __syncthreads();
  float S = 0.f, S2 = 0.f;
#pragma unroll
  for (int i = 0; i < 8; ++i) { S += rs[i]; S2 += rs2[i]; }
  const float mean = S / dm;
  const float inv  = rsqrtf(S2 / dm - mean * mean + 1e-5f);
  for (int i = 0; i < per; ++i) {
    const int idx = threadIdx.x + (i << 8);
    out[(int64_t)row * dm + idx] = (vals[i] - mean) * inv * w[idx] + bias[idx];
  }
}

// In-place softmax over 256 cols; rows padded to 16 per batch sample.
__global__ __launch_bounds__(256) void softmax_ca(float* __restrict__ X)
{
  const int b = blockIdx.x >> 3;
  const int h = blockIdx.x & 7;
  float* p = X + ((int64_t)b * 16 + h) * 256;
  const int s = threadIdx.x;
  const float x = p[s];
  __shared__ float red[8];
  float m = warp_max(x);
  if ((s & 31) == 0) red[s >> 5] = m;
  __syncthreads();
  float bmax = red[0];
#pragma unroll
  for (int i = 1; i < 8; ++i) bmax = fmaxf(bmax, red[i]);
  __syncthreads();
  const float e = __expf(x - bmax);
  float z = warp_sum(e);
  if ((s & 31) == 0) red[s >> 5] = z;
  __syncthreads();
  float Z = 0.f;
#pragma unroll
  for (int i = 0; i < 8; ++i) Z += red[i];
  p[s] = e / Z;
}

__global__ void rownorm_inv_bf16(const bf16_t* __restrict__ X,
                                 float* __restrict__ out, int rows, int cols) {
  const int row = blockIdx.x * 8 + (threadIdx.x >> 5);
  if (row >= rows) return;
  const bf16_t* p = X + (int64_t)row * cols;
  float s = 0.f;
  for (int i = threadIdx.x & 31; i < cols; i += 32) {
    const float v = (float)p[i];
    s = fmaf(v, v, s);
  }
  s = warp_sum(s);
  if ((threadIdx.x & 31) == 0) out[row] = rsqrtf(s);
}

__global__ void rownorm_inv_f32(const float* __restrict__ X,
                                float* __restrict__ out, int rows, int cols) {
  const int row = blockIdx.x * 8 + (threadIdx.x >> 5);
  if (row >= rows) return;
  const float* p = X + (int64_t)row * cols;
  float s = 0.f;
  for (int i = threadIdx.x & 31; i < cols; i += 32) s = fmaf(p[i], p[i], s);
  s = warp_sum(s);
  if ((threadIdx.x & 31) == 0) out[row] = rsqrtf(s);
}

// A2 (256x1536 bf16): rows 0..127 = q~ (pre-scaled), rows 128..255 = x_sq.
__global__ void pack_a2(const float* __restrict__ qtil,
                        const float* __restrict__ xsq,
                        bf16_t* __restrict__ A2) {
  const int idx = blockIdx.x * 256 + threadIdx.x;   // 256*1536 total
  const int r = idx / DMODEL, c = idx % DMODEL;
  const float v = (r < 128) ? qtil[r * DMODEL + c] : xsq[(r - 128) * DMODEL + c];
  A2[idx] = (bf16_t)v;
}

// Final: softmax over s of scores[i,j,:], weighted sum of dot*invm, scale.
__global__ __launch_bounds__(256) void logits_reduce(
    const float* __restrict__ sc, const float* __restrict__ dt,
    const float* __restrict__ invm, const float* __restrict__ invx,
    const float* __restrict__ logit_scale, float* __restrict__ out)
{
  const int pair = blockIdx.x;
  const int i = pair >> 7, j = pair & 127;
  const int s = threadIdx.x;
  const float x = sc[(int64_t)pair * 256 + s];
  const float d = dt[(int64_t)pair * 256 + s];
  const float im = invm[j * 256 + s];
  __shared__ float red[8];
  float m = warp_max(x);
  if ((s & 31) == 0) red[s >> 5] = m;
  __syncthreads();
  float bmax = red[0];
#pragma unroll
  for (int k = 1; k < 8; ++k) bmax = fmaxf(bmax, red[k]);
  __syncthreads();
  const float e = __expf(x - bmax);
  float z = warp_sum(e);
  if ((s & 31) == 0) red[s >> 5] = z;
  __syncthreads();
  float Z = 0.f;
#pragma unroll
  for (int k = 0; k < 8; ++k) Z += red[k];
  __syncthreads();
  float num = warp_sum(e * d * im);
  if ((s & 31) == 0) red[s >> 5] = num;
  __syncthreads();
  float Nm = 0.f;
#pragma unroll
  for (int k = 0; k < 8; ++k) Nm += red[k];
  if (threadIdx.x == 0) {
    const float l = __expf(*logit_scale) * invx[i] * Nm / Z;
    out[i * 128 + j] = l;                 // logits_per_table
    out[128 * 128 + j * 128 + i] = l;     // transpose
  }
}

// ------------------------------ host driver --------------------------------

extern "C" void kernel_launch(void* const* d_in, const int* in_sizes, int n_in,
                              void* d_out, int out_size, void* d_ws, size_t ws_size,
                              hipStream_t stream)
{
  (void)in_sizes; (void)n_in; (void)out_size; (void)ws_size;
  const float* x         = (const float*)d_in[0];
  const float* memory    = (const float*)d_in[1];
  const float* emb_w     = (const float*)d_in[2];
  const float* emb_b     = (const float*)d_in[3];
  const float* sa_in_w   = (const float*)d_in[4];
  const float* sa_in_b   = (const float*)d_in[5];
  const float* sa_out_w  = (const float*)d_in[6];
  const float* sa_out_b  = (const float*)d_in[7];
  const float* ca_in_w   = (const float*)d_in[8];
  const float* ca_in_b   = (const float*)d_in[9];
  const float* ca_out_w  = (const float*)d_in[10];
  const float* ca_out_b  = (const float*)d_in[11];
  const float* lin1_w    = (const float*)d_in[12];
  const float* lin1_b    = (const float*)d_in[13];
  const float* lin2_w    = (const float*)d_in[14];
  const float* lin2_b    = (const float*)d_in[15];
  const float* ln1_w     = (const float*)d_in[16];
  const float* ln1_b     = (const float*)d_in[17];
  const float* ln2_w     = (const float*)d_in[18];
  const float* ln2_b     = (const float*)d_in[19];
  const float* ln3_w     = (const float*)d_in[20];
  const float* ln3_b     = (const float*)d_in[21];
  const float* fin_w     = (const float*)d_in[22];
  const float* fin_b     = (const float*)d_in[23];
  const float* attn_in_w = (const float*)d_in[24];
  const float* attn_in_b = (const float*)d_in[25];
  const float* logit_sc  = (const float*)d_in[26];

  const int Bt = BATCH, S = SEQ, D = DMODEL, H = NHEAD, DH = DHEAD, FF = DFF, Di = DIN;

  char* ws = (char*)d_ws;
  size_t off = 0;
  auto alloc = [&](size_t bytes) -> char* {
    char* p = ws + off;
    off += (bytes + 255) & ~(size_t)255;
    return p;
  };
  bf16_t* memB    = (bf16_t*)alloc((size_t)Bt * S * D * 2);
  float*  scoresB = (float*) alloc((size_t)Bt * Bt * S * 4);
  float*  dotsB   = (float*) alloc((size_t)Bt * Bt * S * 4);
  float*  rbuf    = (float*) alloc((size_t)Bt * 16 * D * 4);
  float*  cbuf    = (float*) alloc((size_t)Bt * 16 * D * 4);
  float*  attnb   = (float*) alloc((size_t)Bt * 16 * S * 4);
  float*  t0      = (float*) alloc((size_t)Bt * D * 4);
  float*  sv      = (float*) alloc((size_t)Bt * D * 4);
  float*  osa     = (float*) alloc((size_t)Bt * D * 4);
  float*  t1      = (float*) alloc((size_t)Bt * D * 4);
  float*  qpca    = (float*) alloc((size_t)Bt * D * 4);
  float*  oca     = (float*) alloc((size_t)Bt * D * 4);
  float*  o2      = (float*) alloc((size_t)Bt * D * 4);
  float*  t2      = (float*) alloc((size_t)Bt * D * 4);
  float*  f1      = (float*) alloc((size_t)Bt * FF * 4);
  float*  f2      = (float*) alloc((size_t)Bt * D * 4);
  float*  t3      = (float*) alloc((size_t)Bt * D * 4);
  float*  qp      = (float*) alloc((size_t)Bt * D * 4);
  float*  qtil    = (float*) alloc((size_t)Bt * D * 4);
  bf16_t* A2      = (bf16_t*)alloc((size_t)2 * Bt * D * 2);
  float*  invm    = (float*) alloc((size_t)Bt * S * 4);
  float*  invx    = (float*) alloc((size_t)Bt * 4);

  const float inv_dh = 1.0f / sqrtf((float)DH);
  const float inv_d  = 1.0f / sqrtf((float)D);

  // memory -> bf16 once (halves the dominant HBM traffic, feeds WMMA).
  {
    const int64_t n = (int64_t)Bt * S * D;
    f32_to_bf16<<<dim3((unsigned)((n + 255) / 256)), 256, 0, stream>>>(memory, memB, n);
  }
  {
    const int64_t n = (int64_t)Bt * 16 * D;   // zero r-pad rows (8..15)
    zero_f32<<<dim3((unsigned)((n + 255) / 256)), 256, 0, stream>>>(rbuf, n);
  }

  // t0 = x @ emb_w^T + emb_b
  gemm_wmma<<<dim3(D / 16), 256, 0, stream>>>(x, Di, emb_w, Di, 0, t0, D, emb_b,
                                              Bt, D, Di, 1.f, 0);
  // self-attn with L=1: softmax==1 -> o = (t0 Wv^T + bv) Wo^T + bo
  gemm_wmma<<<dim3(D / 16), 256, 0, stream>>>(t0, D, sa_in_w + (size_t)2 * D * D, D, 0,
                                              sv, D, sa_in_b + 2 * D, Bt, D, D, 1.f, 0);
  gemm_wmma<<<dim3(D / 16), 256, 0, stream>>>(sv, D, sa_out_w, D, 0, osa, D, sa_out_b,
                                              Bt, D, D, 1.f, 0);
  ln_residual<<<dim3(Bt), 256, 0, stream>>>(t0, osa, ln1_w, ln1_b, t1, D);

  // cross-attention, K/V projections folded through the attention math.
  gemm_wmma<<<dim3(D / 16), 256, 0, stream>>>(t1, D, ca_in_w, D, 0, qpca, D, ca_in_b,
                                              Bt, D, D, 1.f, 0);
  for (int h = 0; h < H; ++h)   // r[b,h,:] = Wk_h^T qp_h / sqrt(dh)
    gemm_wmma<<<dim3(D / 16), 256, 0, stream>>>(
        qpca + h * DH, D, ca_in_w + (size_t)(D + h * DH) * D, D, 1,
        rbuf + h * D, 16 * D, nullptr, Bt, D, DH, inv_dh, 0);
  gemm16_batched<<<dim3(Bt, S / 128), 256, 0, stream>>>(   // scores[b,h,s]
      rbuf, (int64_t)16 * D, D, D, memB, (int64_t)S * D, D, 0,
      attnb, (int64_t)16 * S, S, S, 1.f);
  softmax_ca<<<dim3(Bt * H), 256, 0, stream>>>(attnb);
  gemm16_batched<<<dim3(Bt, D / 128), 256, 0, stream>>>(   // c[b,h,:] = attn @ mem
      attnb, (int64_t)16 * S, S, S, memB, (int64_t)S * D, D, 1,
      cbuf, (int64_t)16 * D, D, D, 1.f);
  for (int h = 0; h < H; ++h)   // o[b,h] = Wv_h c[b,h] + bv_h
    gemm_wmma<<<dim3(DH / 16), 256, 0, stream>>>(
        cbuf + h * D, 16 * D, ca_in_w + (size_t)(2 * D + h * DH) * D, D, 0,
        oca + h * DH, D, ca_in_b + 2 * D + h * DH, Bt, DH, D, 1.f, 0);
  gemm_wmma<<<dim3(D / 16), 256, 0, stream>>>(oca, D, ca_out_w, D, 0, o2, D, ca_out_b,
                                              Bt, D, D, 1.f, 0);
  ln_residual<<<dim3(Bt), 256, 0, stream>>>(t1, o2, ln2_w, ln2_b, t2, D);

  // feed-forward
  gemm_wmma<<<dim3(FF / 16), 256, 0, stream>>>(t2, D, lin1_w, D, 0, f1, FF, lin1_b,
                                               Bt, FF, D, 1.f, 1);
  gemm_wmma<<<dim3(D / 16), 256, 0, stream>>>(f1, FF, lin2_w, FF, 0, f2, D, lin2_b,
                                              Bt, D, FF, 1.f, 0);
  ln_residual<<<dim3(Bt), 256, 0, stream>>>(t2, f2, ln3_w, ln3_b, t3, D);

  float* xsq = (float*)d_out + 2 * Bt * Bt;   // third output, also reused below
  ln_residual<<<dim3(Bt), 256, 0, stream>>>(t3, nullptr, fin_w, fin_b, xsq, D);

  // folded final attention: q~ = (xsq Wq^T + bq) Wk / sqrt(D)
  gemm_wmma<<<dim3(D / 16), 256, 0, stream>>>(xsq, D, attn_in_w, D, 0, qp, D,
                                              attn_in_b, Bt, D, D, 1.f, 0);
  gemm_wmma<<<dim3(D / 16), 256, 0, stream>>>(qp, D, attn_in_w + (size_t)D * D, D, 1,
                                              qtil, D, nullptr, Bt, D, D, inv_d, 0);
  pack_a2<<<dim3(2 * Bt * D / 256), 256, 0, stream>>>(qtil, xsq, A2);
  rownorm_inv_bf16<<<dim3(Bt * S / 8), 256, 0, stream>>>(memB, invm, Bt * S, D);
  rownorm_inv_f32<<<dim3(Bt / 8), 256, 0, stream>>>(xsq, invx, Bt, D);

  attn_scores_dots<<<dim3(4, Bt), 256, 0, stream>>>(A2, memB, scoresB, dotsB, D);
  logits_reduce<<<dim3(Bt * Bt), 256, 0, stream>>>(scoresB, dotsB, invm, invx,
                                                   logit_sc, (float*)d_out);
}
